// RGCNLayer_53094385713806
// MI455X (gfx1250) — matrix operand order repaired and checked by reading, not compiled, not run on hardware
//
#include <hip/hip_runtime.h>

typedef __attribute__((ext_vector_type(16))) __bf16   v16bf;
typedef __attribute__((ext_vector_type(8)))  float    v8f;
typedef __attribute__((ext_vector_type(8)))  unsigned int v8u;
typedef unsigned int u32;
typedef unsigned short u16;

#define N_NODES 50000
#define N_EDGES 1600000
#define FEAT    48
#define KP      64          // K padded to 2x32 for 16x16x32 WMMA
#define NREL    8
#define NTILES  (N_NODES/16)   // 3125 exactly

// ---------------- workspace layout (bytes) ----------------
#define OFF_W1B   ((size_t)0)                         // 12288 u32 = 49152 B
#define OFF_W2B   ((size_t)49152)                     // 49152 B
#define OFF_H2    ((size_t)98304)                     // 50000*8*48*4 = 76,800,000 B
#define OFF_CNT   ((size_t)76898304)                  // 50000*4
#define OFF_ROW   ((size_t)77098304)                  // 50001*4
#define OFF_CUR   ((size_t)77298308)                  // 50000*4
#define OFF_EID   ((size_t)77498308)                  // 1600000*4 -> ends ~83.9 MB

__device__ __forceinline__ u16 f2bf(float x) {
  u32 u = __float_as_uint(x);
  u += 0x7fffu + ((u >> 16) & 1u);   // round to nearest even
  return (u16)(u >> 16);
}
__device__ __forceinline__ u32 pack2bf(float a, float b) {
  return (u32)f2bf(a) | ((u32)f2bf(b) << 16);
}

// ---- convert weights to bf16, K padded to 64, layout [r][K][N] as dword pairs ----
__global__ void k_convert_w(const float* __restrict__ w1, const float* __restrict__ w2,
                            u32* __restrict__ w1b, u32* __restrict__ w2b) {
  int idx = blockIdx.x * blockDim.x + threadIdx.x;
  const int TOT = NREL * KP * (FEAT / 2);   // 12288
  if (idx >= TOT) return;
  int r = idx / ((KP / 2) * FEAT);
  int k = 2 * ((idx / FEAT) % (KP / 2));
  int c = idx % FEAT;
  float a1 = 0.f, b1 = 0.f, a2 = 0.f, b2 = 0.f;
  if (k < FEAT) {
    const float* p1 = w1 + ((size_t)(r * FEAT + k) * FEAT + c);
    a1 = p1[0]; b1 = p1[FEAT];
    const float* p2 = w2 + ((size_t)(r * FEAT + k) * FEAT + c);
    a2 = p2[0]; b2 = p2[FEAT];
  }
  w1b[idx] = pack2bf(a1, b1);
  w2b[idx] = pack2bf(a2, b2);
}

// ---- CSR build: histogram ----
__global__ void k_hist(const int* __restrict__ dst, int* __restrict__ cnt) {
  int e = blockIdx.x * blockDim.x + threadIdx.x;
  if (e < N_EDGES) atomicAdd(&cnt[dst[e]], 1);
}

// ---- CSR build: single-block exclusive scan over counts ----
__global__ void k_scan(const int* __restrict__ cnt, int* __restrict__ rowptr,
                       int* __restrict__ cursor) {
  __shared__ int s[1024];
  __shared__ int carry_s;
  int t = threadIdx.x;
  if (t == 0) carry_s = 0;
  __syncthreads();
  for (int base = 0; base < N_NODES; base += 1024) {
    int i = base + t;
    int x = (i < N_NODES) ? cnt[i] : 0;
    s[t] = x;
    __syncthreads();
    for (int off = 1; off < 1024; off <<= 1) {
      int v = 0;
      if (t >= off) v = s[t - off];
      __syncthreads();
      s[t] += v;
      __syncthreads();
    }
    int incl = s[t];
    int excl = incl - x;
    int carry = carry_s;
    if (i < N_NODES) { rowptr[i] = carry + excl; cursor[i] = carry + excl; }
    __syncthreads();
    if (t == 1023) carry_s = carry + incl;
    __syncthreads();
  }
  if (t == 0) rowptr[N_NODES] = carry_s;
}

// ---- CSR build: fill edge ids ----
__global__ void k_fill(const int* __restrict__ dst, int* __restrict__ cursor,
                       int* __restrict__ eids) {
  int e = blockIdx.x * blockDim.x + threadIdx.x;
  if (e < N_EDGES) {
    int p = atomicAdd(&cursor[dst[e]], 1);
    eids[p] = e;
  }
}

// ---- WMMA: h2[n,r,:] = relu(x[n] @ W1[r] + b1) @ W2[r] ----
// block = 256 threads = 8 waves; wave w handles relation w for a 16-node tile.
__global__ __launch_bounds__(256) void k_h2(
    const float* __restrict__ x, const u32* __restrict__ W1, const u32* __restrict__ W2,
    const float* __restrict__ bias1, float* __restrict__ h2) {
  __shared__ u32 sIn[16 * 32];               // 16 rows x 64 K-vals as bf16 pairs
  __shared__ u16 sH1[NREL][16 * KP];         // per-wave h1 staging (bf16)

  const int tid  = threadIdx.x;
  const int wave = tid >> 5;                 // == relation
  const int lane = tid & 31;
  const int r    = wave;
  const int ln16 = lane & 15;                // A-row / CD-column
  const int hi   = lane >> 4;

  // B fragments (weights) for this wave's relation, cached in VGPRs.
  // B 32x16 bf16 layout: lane column N = ln16; VGPR v holds the same K pair as A's VGPR v.
  v8u bw1[3][2], bw2[3][2];
  #pragma unroll
  for (int nt = 0; nt < 3; ++nt)
    #pragma unroll
    for (int kc = 0; kc < 2; ++kc)
      #pragma unroll
      for (int v = 0; v < 8; ++v) {
        int pr = (v < 4 ? v : v + 4) + hi * 4 + 16 * kc;
        int di = (r * (KP / 2) + pr) * FEAT + nt * 16 + ln16;
        bw1[nt][kc][v] = W1[di];
        bw2[nt][kc][v] = W2[di];
      }
  float b1v[3];
  #pragma unroll
  for (int nt = 0; nt < 3; ++nt) b1v[nt] = bias1[nt * 16 + ln16];

  // zero the K-padding columns (48..63) of this wave's sH1 once
  {
    u16* p = &sH1[wave][ln16 * KP + 48 + hi * 8];
    #pragma unroll
    for (int j = 0; j < 8; ++j) p[j] = 0;
  }

  for (int tile = blockIdx.x; tile < NTILES; tile += gridDim.x) {
    const int nb = tile * 16;
    __syncthreads();   // previous iteration done with sIn / sH1
    // cooperative load of 16x48 input tile -> bf16, K padded to 64
    for (int idx = tid; idx < 16 * 32; idx += 256) {
      int row = idx >> 5, c = idx & 31;
      int k0 = 2 * c;
      float a = 0.f, b = 0.f;
      if (k0 < FEAT) {
        const float* p = x + (size_t)(nb + row) * FEAT + k0;
        a = p[0]; b = p[1];
      }
      sIn[idx] = pack2bf(a, b);
    }
    __syncthreads();

    // A fragments (16x32 bf16): lane row = ln16; VGPR v holds K pair
    // v<4 -> K = 2v + 8*hi ; v>=4 -> K = 2v + 8 + 8*hi  (pair index = K/2)
    v8u a0u, a1u;
    #pragma unroll
    for (int v = 0; v < 8; ++v) {
      int pr = (v < 4 ? v : v + 4) + hi * 4;
      a0u[v] = sIn[ln16 * 32 + pr];
      a1u[v] = sIn[ln16 * 32 + pr + 16];
    }
    v16bf A0 = __builtin_bit_cast(v16bf, a0u);
    v16bf A1 = __builtin_bit_cast(v16bf, a1u);

    // ---- stage 1: h1 = relu(x @ W1[r] + b1) -> sH1 (bf16) ----
    #pragma unroll
    for (int nt = 0; nt < 3; ++nt) {
      v8f acc = {};
      acc = __builtin_amdgcn_wmma_f32_16x16x32_bf16(
          false, A0, false, __builtin_bit_cast(v16bf, bw1[nt][0]), (short)0, acc, false, false);
      acc = __builtin_amdgcn_wmma_f32_16x16x32_bf16(
          false, A1, false, __builtin_bit_cast(v16bf, bw1[nt][1]), (short)0, acc, false, false);
      // C/D layout: column N = ln16 (+nt*16), row M = p + 8*hi
      #pragma unroll
      for (int p = 0; p < 8; ++p) {
        float t = acc[p] + b1v[nt];
        t = t > 0.f ? t : 0.f;
        sH1[wave][(p + 8 * hi) * KP + nt * 16 + ln16] = f2bf(t);
      }
    }
    __syncthreads();   // ordering barrier: u16 stores -> u32 reloads

    // A fragments of h1 from this wave's staging buffer
    const u32* H = (const u32*)&sH1[wave][0];   // KP shorts = 32 dwords per row
    v8u c0u, c1u;
    #pragma unroll
    for (int v = 0; v < 8; ++v) {
      int pr = (v < 4 ? v : v + 4) + hi * 4;
      c0u[v] = H[ln16 * 32 + pr];
      c1u[v] = H[ln16 * 32 + pr + 16];
    }
    v16bf C0 = __builtin_bit_cast(v16bf, c0u);
    v16bf C1 = __builtin_bit_cast(v16bf, c1u);

    // ---- stage 2: h2 = h1 @ W2[r] -> global (f32) ----
    #pragma unroll
    for (int nt = 0; nt < 3; ++nt) {
      v8f acc = {};
      acc = __builtin_amdgcn_wmma_f32_16x16x32_bf16(
          false, C0, false, __builtin_bit_cast(v16bf, bw2[nt][0]), (short)0, acc, false, false);
      acc = __builtin_amdgcn_wmma_f32_16x16x32_bf16(
          false, C1, false, __builtin_bit_cast(v16bf, bw2[nt][1]), (short)0, acc, false, false);
      #pragma unroll
      for (int p = 0; p < 8; ++p) {
        int node = nb + p + 8 * hi;
        h2[(size_t)(node * NREL + r) * FEAT + nt * 16 + ln16] = acc[p];
      }
    }
  }
}

// ---- aggregation: 16 lanes per dst node, CSR edge list, + bias2, relu ----
__global__ __launch_bounds__(256) void k_agg(
    const float* __restrict__ h2, const int* __restrict__ rowptr,
    const int* __restrict__ eids, const int* __restrict__ src,
    const int* __restrict__ rel, const float* __restrict__ norm,
    const float* __restrict__ bias2, float* __restrict__ out) {
  int tid = threadIdx.x;
  int g = tid >> 4;
  int l = tid & 15;
  int n = blockIdx.x * 16 + g;
  if (n >= N_NODES) return;
  float a0 = 0.f, a1 = 0.f, a2 = 0.f;
  int beg = rowptr[n], end = rowptr[n + 1];
  for (int j = beg; j < end; ++j) {
    int e = eids[j];
    int s = src[e];
    int rr = rel[e];
    float nv = norm[e];
    const float* m = h2 + (size_t)(s * NREL + rr) * FEAT;
    a0 += m[l]      * nv;
    a1 += m[l + 16] * nv;
    a2 += m[l + 32] * nv;
  }
  a0 += bias2[l];
  a1 += bias2[l + 16];
  a2 += bias2[l + 32];
  float* o = out + (size_t)n * FEAT;
  o[l]      = a0 > 0.f ? a0 : 0.f;
  o[l + 16] = a1 > 0.f ? a1 : 0.f;
  o[l + 32] = a2 > 0.f ? a2 : 0.f;
}

extern "C" void kernel_launch(void* const* d_in, const int* in_sizes, int n_in,
                              void* d_out, int out_size, void* d_ws, size_t ws_size,
                              hipStream_t stream) {
  const float* inputs = (const float*)d_in[0];
  const float* norm   = (const float*)d_in[1];
  const float* w1     = (const float*)d_in[2];
  const float* w2     = (const float*)d_in[3];
  const float* b1     = (const float*)d_in[4];
  const float* b2     = (const float*)d_in[5];
  const int*   src    = (const int*)d_in[6];
  const int*   dst    = (const int*)d_in[7];
  const int*   rel    = (const int*)d_in[8];

  char* ws = (char*)d_ws;
  u32*   W1B    = (u32*)(ws + OFF_W1B);
  u32*   W2B    = (u32*)(ws + OFF_W2B);
  float* H2     = (float*)(ws + OFF_H2);
  int*   CNT    = (int*)(ws + OFF_CNT);
  int*   ROW    = (int*)(ws + OFF_ROW);
  int*   CUR    = (int*)(ws + OFF_CUR);
  int*   EID    = (int*)(ws + OFF_EID);

  hipMemsetAsync(CNT, 0, (size_t)N_NODES * 4, stream);
  k_convert_w<<<48, 256, 0, stream>>>(w1, w2, W1B, W2B);
  k_hist<<<(N_EDGES + 255) / 256, 256, 0, stream>>>(dst, CNT);
  k_scan<<<1, 1024, 0, stream>>>(CNT, ROW, CUR);
  k_fill<<<(N_EDGES + 255) / 256, 256, 0, stream>>>(dst, CUR, EID);
  k_h2<<<1024, 256, 0, stream>>>(inputs, W1B, W2B, b1, H2);
  k_agg<<<NTILES, 256, 0, stream>>>(H2, ROW, EID, src, rel, norm, b2, (float*)d_out);
}
